// Decoder_72258529788352
// MI455X (gfx1250) — matrix operand order repaired and checked by reading, not compiled, not run on hardware
//
#include <hip/hip_runtime.h>
#include <hip/hip_bf16.h>

#define H    1024
#define BSZ  2048
#define TSZ  96
#define XK   64   // LSTM input dim padded 60 -> 64 (two K=32 WMMA chunks)

#define AS1 __attribute__((address_space(1)))
#define AS3 __attribute__((address_space(3)))

#if __has_builtin(__builtin_amdgcn_global_load_async_to_lds_b128)
#define HAVE_ASYNC_LDS 1
#else
#define HAVE_ASYNC_LDS 0
#endif

typedef __attribute__((ext_vector_type(16))) __bf16 v16bf;
typedef __attribute__((ext_vector_type(8)))  float  v8f;
typedef int v4i __attribute__((vector_size(16)));   // matches builtin's int4 param type

union ABFrag { uint4 u[2]; v16bf v; };

__device__ __forceinline__ float sigf(float x) { return 1.0f / (1.0f + __expf(-x)); }

// copy one 16B chunk global -> LDS (async-to-LDS path, tracked on ASYNCcnt)
__device__ __forceinline__ void stage16(const __bf16* gsrc, __bf16* ldst) {
#if HAVE_ASYNC_LDS
  __builtin_amdgcn_global_load_async_to_lds_b128(
      (AS1 v4i*)gsrc, (AS3 v4i*)ldst, 0, 0);
#else
  *reinterpret_cast<uint4*>(ldst) = *reinterpret_cast<const uint4*>(gsrc);
#endif
}

__device__ __forceinline__ void stage_wait() {
#if HAVE_ASYNC_LDS
#if __has_builtin(__builtin_amdgcn_s_wait_asynccnt)
  __builtin_amdgcn_s_wait_asynccnt(0);
#else
  asm volatile("s_wait_asynccnt 0x0" ::: "memory");
#endif
#endif
}

// ---------------- prologue kernels ----------------

__global__ void cvt_bf16_kernel(const float* __restrict__ src, __bf16* __restrict__ dst, int n) {
  int i = blockIdx.x * blockDim.x + threadIdx.x;
  if (i < n) dst[i] = (__bf16)src[i];
}

// W_ih0 [4096][60] fp32 -> [4096][64] bf16 (zero-padded K)
__global__ void pad_ih0_kernel(const float* __restrict__ src, __bf16* __restrict__ dst) {
  int i = blockIdx.x * blockDim.x + threadIdx.x;   // 4096*64 threads
  int r = i >> 6, k = i & 63;
  dst[i] = (k < 60) ? (__bf16)src[r * 60 + k] : (__bf16)0.0f;
}

__global__ void init_state_kernel(const float* __restrict__ enc_h, const float* __restrict__ enc_c,
                                  __bf16* __restrict__ h1, float* __restrict__ c1,
                                  __bf16* __restrict__ h2, float* __restrict__ c2) {
  int i = blockIdx.x * blockDim.x + threadIdx.x;   // B*H threads
  h1[i] = (__bf16)enc_h[i];
  c1[i] = enc_c[i];
  h2[i] = (__bf16)enc_h[(size_t)BSZ * H + i];
  c2[i] = enc_c[(size_t)BSZ * H + i];
}

// Build teacher-forced input x[T][B][64] in bf16:
// [0]=prev_y, [1..32]=dec_known, [33..43]=cat embeddings, [44..59]=group emb, [60..63]=0
__global__ void build_x_kernel(const float* __restrict__ dec_known, const float* __restrict__ target_y,
                               const float* __restrict__ last_enc, const int* __restrict__ gids,
                               const int* __restrict__ cprov, const int* __restrict__ ccust,
                               const int* __restrict__ cprice, const int* __restrict__ clevel,
                               const float* __restrict__ g_emb, const float* __restrict__ e_prov,
                               const float* __restrict__ e_cust, const float* __restrict__ e_price,
                               const float* __restrict__ e_level, __bf16* __restrict__ x) {
  int i = blockIdx.x * blockDim.x + threadIdx.x;   // B*T threads
  if (i >= BSZ * TSZ) return;
  int b = i / TSZ, t = i % TSZ;
  __bf16* row = x + ((size_t)t * BSZ + b) * XK;
  row[0] = (__bf16)((t == 0) ? last_enc[b] : target_y[(size_t)b * TSZ + t - 1]);
  const float* dk = dec_known + ((size_t)b * TSZ + t) * 32;
  #pragma unroll
  for (int j = 0; j < 32; ++j) row[1 + j] = (__bf16)dk[j];
  const float* ep = e_prov + cprov[b] * 4;
  #pragma unroll
  for (int j = 0; j < 4; ++j) row[33 + j] = (__bf16)ep[j];
  const float* ec = e_cust + ccust[b] * 2;
  #pragma unroll
  for (int j = 0; j < 2; ++j) row[37 + j] = (__bf16)ec[j];
  const float* epr = e_price + cprice[b] * 3;
  #pragma unroll
  for (int j = 0; j < 3; ++j) row[39 + j] = (__bf16)epr[j];
  const float* el = e_level + clevel[b] * 2;
  #pragma unroll
  for (int j = 0; j < 2; ++j) row[42 + j] = (__bf16)el[j];
  const float* gv = g_emb + gids[b] * 16;
  #pragma unroll
  for (int j = 0; j < 16; ++j) row[44 + j] = (__bf16)gv[j];
  row[60] = row[61] = row[62] = row[63] = (__bf16)0.0f;
}

// ---------------- fused LSTM cell step ----------------
// G[b, n + g*H] = A1[b,:] @ W1[n+g*H,:]^T + A2[b,:] @ W2[n+g*H,:]^T   (bf16 WMMA, f32 acc)
// then in-register: c = sig(f)*c + sig(i)*tanh(g); h = sig(o)*tanh(c)
// Block: 256 threads = 8 waves; tile = 128 batch rows x 32 n-cols x all 4 gates.
// Wave w owns m rows [w*16, w*16+16); computes 2 n-subtiles x 4 gates = 8 WMMA tiles.
// Both A (128x32) and B (4 gates x 32 cols x 32 K) staged through LDS each K-chunk.
__global__ __launch_bounds__(256)
void lstm_cell_kernel(const __bf16* __restrict__ A1, int lda1,
                      const __bf16* __restrict__ W1, int K1,
                      const __bf16* __restrict__ A2,
                      const __bf16* __restrict__ W2,
                      const float* __restrict__ bias,
                      float* __restrict__ c,
                      __bf16* __restrict__ h_out) {
  __shared__ __bf16 smemA[128 * 32];      // 8 KB batch tile
  __shared__ __bf16 smemB[128 * 32];      // 8 KB weight tile: [gate*32+col][k]
  const int lane = threadIdx.x & 31;
  const int wave = threadIdx.x >> 5;
  const int n0 = blockIdx.x * 32;
  const int m0 = blockIdx.y * 128;

  v8f acc[2][4];
  #pragma unroll
  for (int nt = 0; nt < 2; ++nt)
    #pragma unroll
    for (int g = 0; g < 4; ++g)
      acc[nt][g] = (v8f){0.f, 0.f, 0.f, 0.f, 0.f, 0.f, 0.f, 0.f};

  const __bf16* Aarr[2] = {A1, A2};
  const __bf16* Warr[2] = {W1, W2};
  const int Karr[2]  = {K1, H};
  const int ldarr[2] = {lda1, H};

  const int mrow = wave * 16 + (lane & 15);   // A frag row within tile
  const int ka   = (lane < 16) ? 0 : 8;       // A layout: lanes 0-15 hold K 0..7 / 16..23
  const int kb   = (lane < 16) ? 0 : 8;       // B layout: lanes 0-15 hold K 0..7 / 16..23

  for (int ph = 0; ph < 2; ++ph) {
    const __bf16* A = Aarr[ph];
    const __bf16* W = Warr[ph];
    const int K = Karr[ph];
    const int lda = ldarr[ph];
    for (int k0 = 0; k0 < K; k0 += 32) {
      __syncthreads();
      // cooperative stage: 2 b128 of A + 2 b128 of B per thread (async-to-LDS)
      #pragma unroll
      for (int j = 0; j < 2; ++j) {
        const int off = (threadIdx.x + 256 * j) * 8;   // element offset in 128x32 tile
        const int row = off >> 5;
        const int ko  = off & 31;
        stage16(&A[(size_t)(m0 + row) * lda + k0 + ko], &smemA[off]);
        const int gate = row >> 5;                     // B: row -> gate, col
        const int col  = row & 31;
        stage16(&W[(size_t)(gate * H + n0 + col) * K + k0 + ko], &smemB[off]);
      }
      stage_wait();
      __syncthreads();
      // A fragment (16x32 bf16) from LDS
      ABFrag fa;
      fa.u[0] = *reinterpret_cast<const uint4*>(&smemA[mrow * 32 + ka]);
      fa.u[1] = *reinterpret_cast<const uint4*>(&smemA[mrow * 32 + ka + 16]);
      #pragma unroll
      for (int nt = 0; nt < 2; ++nt) {
        #pragma unroll
        for (int g = 0; g < 4; ++g) {
          ABFrag fb;
          const int base = (g * 32 + nt * 16 + (lane & 15)) * 32 + kb;
          fb.u[0] = *reinterpret_cast<const uint4*>(&smemB[base]);
          fb.u[1] = *reinterpret_cast<const uint4*>(&smemB[base + 16]);
          acc[nt][g] = __builtin_amdgcn_wmma_f32_16x16x32_bf16(
              false, fa.v, false, fb.v, (short)0, acc[nt][g], false, false);
        }
      }
    }
  }

  // fused LSTM cell epilogue: this wave holds i,f,g,o for its (m,n) elements
  #pragma unroll
  for (int nt = 0; nt < 2; ++nt) {
    const int n = n0 + nt * 16 + (lane & 15);
    #pragma unroll
    for (int r = 0; r < 8; ++r) {
      const int m = m0 + wave * 16 + r + ((lane < 16) ? 0 : 8);  // C/D layout: VGPR r -> M=r / r+8
      const size_t idx = (size_t)m * H + n;
      float gi = acc[nt][0][r] + bias[n];
      float gf = acc[nt][1][r] + bias[n + H];
      float gg = acc[nt][2][r] + bias[n + 2 * H];
      float go = acc[nt][3][r] + bias[n + 3 * H];
      float cn = sigf(gf) * c[idx] + sigf(gi) * tanhf(gg);
      c[idx] = cn;
      h_out[idx] = (__bf16)(sigf(go) * tanhf(cn));
    }
  }
}

// y[b, t] = h2[b,:] . W_proj + b_proj   (one wave per batch row)
__global__ void proj_kernel(const __bf16* __restrict__ h2, const float* __restrict__ Wp,
                            const float* __restrict__ bp, float* __restrict__ y, int t) {
  int lane = threadIdx.x & 31;
  int wave = threadIdx.x >> 5;
  int b = blockIdx.x * 8 + wave;
  const __bf16* hr = h2 + (size_t)b * H;
  float s = 0.f;
  #pragma unroll 4
  for (int j = 0; j < H / 32; ++j) {
    int nn = j * 32 + lane;
    s += (float)hr[nn] * Wp[nn];
  }
  #pragma unroll
  for (int o = 16; o > 0; o >>= 1) s += __shfl_xor(s, o, 32);
  if (lane == 0) y[(size_t)b * TSZ + t] = s + bp[0];
}

// ---------------- host launcher ----------------

extern "C" void kernel_launch(void* const* d_in, const int* in_sizes, int n_in,
                              void* d_out, int out_size, void* d_ws, size_t ws_size,
                              hipStream_t stream) {
  const float* dec_known = (const float*)d_in[0];
  const float* target_y  = (const float*)d_in[1];
  const float* enc_h     = (const float*)d_in[2];
  const float* enc_c     = (const float*)d_in[3];
  const float* last_enc  = (const float*)d_in[4];
  const int*   gids      = (const int*)d_in[5];
  const int*   cprov     = (const int*)d_in[6];
  const int*   ccust     = (const int*)d_in[7];
  const int*   cprice    = (const int*)d_in[8];
  const int*   clevel    = (const int*)d_in[9];
  const float* g_emb     = (const float*)d_in[10];
  const float* e_prov    = (const float*)d_in[11];
  const float* e_cust    = (const float*)d_in[12];
  const float* e_price   = (const float*)d_in[13];
  const float* e_level   = (const float*)d_in[14];
  const float* W_ih0     = (const float*)d_in[15];
  const float* W_hh0     = (const float*)d_in[16];
  const float* b0        = (const float*)d_in[17];
  const float* W_ih1     = (const float*)d_in[18];
  const float* W_hh1     = (const float*)d_in[19];
  const float* b1        = (const float*)d_in[20];
  const float* W_proj    = (const float*)d_in[21];
  const float* b_proj    = (const float*)d_in[22];
  float* y = (float*)d_out;

  // workspace carve-up (~81 MB)
  char* p = (char*)d_ws;
  auto alloc = [&](size_t bytes) -> void* {
    void* r = (void*)p;
    p += (bytes + 255) & ~(size_t)255;
    return r;
  };
  __bf16* Wih0_b = (__bf16*)alloc((size_t)4096 * XK * 2);
  __bf16* Whh0_b = (__bf16*)alloc((size_t)4096 * H * 2);
  __bf16* Wih1_b = (__bf16*)alloc((size_t)4096 * H * 2);
  __bf16* Whh1_b = (__bf16*)alloc((size_t)4096 * H * 2);
  __bf16* x      = (__bf16*)alloc((size_t)TSZ * BSZ * XK * 2);
  __bf16* h1a    = (__bf16*)alloc((size_t)BSZ * H * 2);
  __bf16* h1b    = (__bf16*)alloc((size_t)BSZ * H * 2);
  __bf16* h2a    = (__bf16*)alloc((size_t)BSZ * H * 2);
  __bf16* h2b    = (__bf16*)alloc((size_t)BSZ * H * 2);
  float*  c1     = (float*)alloc((size_t)BSZ * H * 4);
  float*  c2     = (float*)alloc((size_t)BSZ * H * 4);

  const int NW = 4096 * H;
  cvt_bf16_kernel<<<NW / 256, 256, 0, stream>>>(W_hh0, Whh0_b, NW);
  cvt_bf16_kernel<<<NW / 256, 256, 0, stream>>>(W_ih1, Wih1_b, NW);
  cvt_bf16_kernel<<<NW / 256, 256, 0, stream>>>(W_hh1, Whh1_b, NW);
  pad_ih0_kernel<<<4096 * XK / 256, 256, 0, stream>>>(W_ih0, Wih0_b);
  build_x_kernel<<<(BSZ * TSZ + 255) / 256, 256, 0, stream>>>(
      dec_known, target_y, last_enc, gids, cprov, ccust, cprice, clevel,
      g_emb, e_prov, e_cust, e_price, e_level, x);
  init_state_kernel<<<BSZ * H / 256, 256, 0, stream>>>(enc_h, enc_c, h1a, c1, h2a, c2);

  dim3 grid(H / 32, BSZ / 128);   // 32 x 16 = 512 blocks per layer step
  for (int t = 0; t < TSZ; ++t) {
    const __bf16* h1_in = (t & 1) ? h1b : h1a;
    __bf16*      h1_out = (t & 1) ? h1a : h1b;
    const __bf16* h2_in = (t & 1) ? h2b : h2a;
    __bf16*      h2_out = (t & 1) ? h2a : h2b;
    // layer 0: x_t (K=64) + h1_prev (K=1024)
    lstm_cell_kernel<<<grid, 256, 0, stream>>>(x + (size_t)t * BSZ * XK, XK, Wih0_b, XK,
                                               h1_in, Whh0_b, b0, c1, h1_out);
    // layer 1: h1_cur (K=1024) + h2_prev (K=1024)
    lstm_cell_kernel<<<grid, 256, 0, stream>>>(h1_out, H, Wih1_b, H,
                                               h2_in, Whh1_b, b1, c2, h2_out);
    proj_kernel<<<BSZ / 8, 256, 0, stream>>>(h2_out, W_proj, b_proj, y, t);
  }
}
